// NaNConv2d_83708912599760
// MI455X (gfx1250) — matrix operand order repaired and checked by reading, not compiled, not run on hardware
//
#include <hip/hip_runtime.h>

// ---------------------------------------------------------------------------
// NaN-aware conv as implicit GEMM on CDNA5 WMMA (bf16 in, f32 accumulate).
//   s1 = conv(x0, K), m = conv(mask, K)  -> same A (weights), two B matrices.
//   out = s1 + patch_mean * m + bias ;  NaN where nan_count >= 576.
// Activations staged in NHWC bf16: im2col tiles contiguous & 16B aligned,
// global->LDS via async b128 (ASYNCcnt), double-buffered across 9 taps.
// Wave tile 32 OC x 32 px: 8 WMMA per 12 LDS b128 reads.
// ---------------------------------------------------------------------------

#define B_   16
#define C_   128
#define H_   128
#define W_   128
#define OC_  256
#define HP   130          // padded H
#define WP   130          // padded W
#define KTOT 1152         // C_*3*3, K ordered (kh,kw,c) with c fastest

typedef __attribute__((ext_vector_type(16))) __bf16 v16bf;
typedef __attribute__((ext_vector_type(8)))  float  v8f;
typedef __attribute__((__vector_size__(16))) int    v4i;

union FragBF { uint4 q[2]; v16bf v; };   // 32B: one wave-fragment slice
union Pack8  { uint4 q;    __bf16 h[8]; };

__device__ __forceinline__ v8f wmma_bf16(v16bf a, v16bf b, v8f c) {
#if defined(__gfx1250__)
  // (neg_a, A, neg_b, B, c_mod, C, reuse_a, reuse_b)
  return __builtin_amdgcn_wmma_f32_16x16x32_bf16(false, a, false, b, (short)0, c,
                                                 false, false);
#else
  (void)a; (void)b;
  return c;
#endif
}

// ---- async global->LDS (CDNA5), with portable fallback --------------------
#if defined(__has_builtin)
#if __has_builtin(__builtin_amdgcn_global_load_async_to_lds_b128) && \
    __has_builtin(__builtin_amdgcn_s_wait_asynccnt)
#define USE_ASYNC_LDS 1
#endif
#endif

__device__ __forceinline__ void cp16(const __bf16* g, __bf16* l) {
#if USE_ASYNC_LDS
  __builtin_amdgcn_global_load_async_to_lds_b128(
      (__attribute__((address_space(1))) v4i*)g,
      (__attribute__((address_space(3))) v4i*)l, 0, 0);
#else
  *(uint4*)l = *(const uint4*)g;
#endif
}

__device__ __forceinline__ void wait_cp() {
#if USE_ASYNC_LDS
  __builtin_amdgcn_s_wait_asynccnt(0);
#endif
}

// ---- pass 1: pad + NaN-split + NCHW->NHWC transpose (via LDS) -------------
__global__ void prep_padT(const float* __restrict__ x,
                          __bf16* __restrict__ x0t, __bf16* __restrict__ mkt) {
  __shared__ __bf16 tx[32 * 132];
  __shared__ __bf16 tm[32 * 132];
  const int bid = blockIdx.x;               // b*130*4 + hp*4 + cc
  const int cc = bid & 3;
  const int hp = (bid >> 2) % HP;
  const int b  = bid / (HP * 4);
  const int c0 = cc * 32;
  const int t  = threadIdx.x;

  for (int e = t; e < 32 * WP; e += 256) {   // coalesced read along w
    int c = e / WP, wp = e % WP;
    float v = 0.0f, m = 0.0f;
    if (hp >= 1 && hp <= H_ && wp >= 1 && wp <= W_) {
      float xv = x[(((size_t)b * C_ + c0 + c) * H_ + (hp - 1)) * W_ + (wp - 1)];
      if (xv != xv) m = 1.0f; else v = xv;
    }
    tx[c * 132 + wp] = (__bf16)v;
    tm[c * 132 + wp] = (__bf16)m;
  }
  __syncthreads();
  for (int e = t; e < WP * 32; e += 256) {   // coalesced write along c
    int wp = e >> 5, c = e & 31;
    size_t o = (((size_t)b * HP + hp) * WP + wp) * C_ + c0 + c;
    x0t[o] = tx[c * 132 + wp];
    mkt[o] = tm[c * 132 + wp];
  }
}

// ---- pass 2: weights -> bf16, K reordered to (kh,kw,c), c fastest ---------
__global__ void prep_wt(const float* __restrict__ kw, __bf16* __restrict__ A) {
  int idx = blockIdx.x * 256 + threadIdx.x;
  if (idx >= OC_ * KTOT) return;
  int k  = idx % KTOT;
  int oc = idx / KTOT;
  int c  = k & 127;
  int kk = k >> 7;
  int kh = kk / 3, kwi = kk % 3;
  A[idx] = (__bf16)kw[(((size_t)oc * C_ + c) * 3 + kh) * 3 + kwi];
}

// ---- pass 3a: per padded pixel, channel sums (NHWC: 128 contiguous) -------
__global__ void stats_cs(const __bf16* __restrict__ x0t, const __bf16* __restrict__ mkt,
                         float* __restrict__ csx, float* __restrict__ csm) {
  int idx = blockIdx.x * 256 + threadIdx.x;
  if (idx >= B_ * HP * WP) return;
  size_t base = (size_t)idx * C_;
  float sx = 0.0f, sm = 0.0f;
  #pragma unroll 4
  for (int i = 0; i < 16; ++i) {
    Pack8 u; u.q = ((const uint4*)(x0t + base))[i];
    #pragma unroll
    for (int j = 0; j < 8; ++j) sx += (float)u.h[j];
  }
  #pragma unroll 4
  for (int i = 0; i < 16; ++i) {
    Pack8 u; u.q = ((const uint4*)(mkt + base))[i];
    #pragma unroll
    for (int j = 0; j < 8; ++j) sm += (float)u.h[j];
  }
  csx[idx] = sx;
  csm[idx] = sm;
}

// ---- pass 3b: 3x3 window sums -> patch_mean, nan_count --------------------
__global__ void stats_patch(const float* __restrict__ csx, const float* __restrict__ csm,
                            float* __restrict__ meanArr, float* __restrict__ ncntArr) {
  int idx = blockIdx.x * 256 + threadIdx.x;
  if (idx >= B_ * H_ * W_) return;
  int ow = idx & 127;
  int oh = (idx >> 7) & 127;
  int b  = idx >> 14;
  float sx = 0.0f, sm = 0.0f;
  #pragma unroll
  for (int kh = 0; kh < 3; ++kh)
    #pragma unroll
    for (int kw = 0; kw < 3; ++kw) {
      int p = ((b * HP) + oh + kh) * WP + (ow + kw);
      sx += csx[p];
      sm += csm[p];
    }
  float valid = (float)KTOT - sm;
  meanArr[idx] = sx / fmaxf(valid, 1.0f);
  ncntArr[idx] = sm;
}

// ---- pass 4: implicit-GEMM conv via WMMA ----------------------------------
// Block = 256 thr = 8 wave32 (4 M x 2 N). Block tile 128 OC x 64 px.
// Wave tile 32 OC x 32 px, both matrices -> 8 accumulators.
// Outer loop: 9 (kh,kw) taps, double-buffered LDS (async prefetch of kk+1).
// Inner: 4 sub-steps of 32 channels; 8 WMMA each -> 32 WMMA/wave/tap.
__global__ __launch_bounds__(256) void conv_wmma(
    const __bf16* __restrict__ x0t, const __bf16* __restrict__ mkt,
    const __bf16* __restrict__ Aw, const float* __restrict__ bias,
    const float* __restrict__ meanArr, const float* __restrict__ ncntArr,
    float* __restrict__ out) {
  // rows padded to 136 bf16 (272B = 17*16B) -> aligned chunks, spread banks
  __shared__ __align__(16) __bf16 sA [2][128 * 136];  // [oc ][c0..127]
  __shared__ __align__(16) __bf16 sBx[2][ 64 * 136];  // [pix][c0..127]
  __shared__ __align__(16) __bf16 sBm[2][ 64 * 136];

  const int t     = threadIdx.x;
  const int lane  = t & 31;
  const int wv    = t >> 5;
  const int waveM = wv & 3;        // 32-OC sub-tile
  const int waveN = wv >> 2;       // 32-px sub-tile

  const int bx     = blockIdx.x;            // 4096 = 2 * 128 * 16
  const int owt    = bx & 1;
  const int oh     = (bx >> 1) & 127;
  const int b      = bx >> 8;
  const int ocBlk  = blockIdx.y * 128;
  const int ocBase = ocBlk + waveM * 32;
  const int owBase = owt * 64 + waveN * 32;

  v8f accS[2][2] = {};   // [mIdx][nIdx]  s1
  v8f accM[2][2] = {};   //               mask-conv

  // cooperative-load roles: thread t owns row (t>>2), 16B column slot (t&3)
  const int lr = t >> 2;               // 0..63
  const int lq = (t & 3) * 8;          // bf16 elements

  const int fm = lane & 15;
  const int g  = lane >> 4;

  // tile loader: tap kk -> buffer p (16 aligned async b128 / thread)
  auto issue_tap = [&](int kk, int p) {
    const int kh = kk / 3, kw = kk % 3;
    {  // A: 128 rows x 256B; thread covers rows lr and lr+64
      const __bf16* gA = Aw + (size_t)(ocBlk + lr) * KTOT + kk * 128 + lq;
      __bf16* lA = &sA[p][lr * 136 + lq];
      #pragma unroll
      for (int i = 0; i < 4; ++i) cp16(gA + i * 32, lA + i * 32);
      const __bf16* gA2 = gA + (size_t)64 * KTOT;
      __bf16* lA2 = lA + 64 * 136;
      #pragma unroll
      for (int i = 0; i < 4; ++i) cp16(gA2 + i * 32, lA2 + i * 32);
    }
    {  // B: 64 px x 256B, both tensors
      const size_t pixbase =
          (((size_t)b * HP + (oh + kh)) * WP + (owt * 64 + kw + lr)) * C_ + lq;
      const __bf16* gx = x0t + pixbase;
      const __bf16* gm = mkt + pixbase;
      __bf16* lx = &sBx[p][lr * 136 + lq];
      __bf16* lm = &sBm[p][lr * 136 + lq];
      #pragma unroll
      for (int i = 0; i < 4; ++i) cp16(gx + i * 32, lx + i * 32);
      #pragma unroll
      for (int i = 0; i < 4; ++i) cp16(gm + i * 32, lm + i * 32);
    }
  };

  issue_tap(0, 0);

  for (int kk = 0; kk < 9; ++kk) {
    const int p = kk & 1;
    wait_cp();
    __syncthreads();                       // buffer p ready for all waves
    if (kk < 8) issue_tap(kk + 1, p ^ 1);  // prefetch into other buffer

    const __bf16* rowA0 = &sA [p][(waveM * 32 + fm) * 136];
    const __bf16* rowA1 = rowA0 + 16 * 136;
    const __bf16* rowB0 = &sBx[p][(waveN * 32 +      fm) * 136];
    const __bf16* rowB1 = &sBx[p][(waveN * 32 + 16 + fm) * 136];
    const __bf16* rowM0 = &sBm[p][(waveN * 32 +      fm) * 136];
    const __bf16* rowM1 = &sBm[p][(waveN * 32 + 16 + fm) * 136];

    #pragma unroll
    for (int cs = 0; cs < 4; ++cs) {
      const int c0 = cs * 32;
      FragBF a0, a1, bx0, bx1, bm0, bm1;
      // A frags: lane half g holds K = g*16 .. g*16+15 contiguous
      a0.q[0] = *(const uint4*)(rowA0 + c0 + g * 16);
      a0.q[1] = *(const uint4*)(rowA0 + c0 + g * 16 + 8);
      a1.q[0] = *(const uint4*)(rowA1 + c0 + g * 16);
      a1.q[1] = *(const uint4*)(rowA1 + c0 + g * 16 + 8);
      // B frags: lane half g holds K = g*16 .. g*16+15 contiguous
      bx0.q[0] = *(const uint4*)(rowB0 + c0 + g * 16);
      bx0.q[1] = *(const uint4*)(rowB0 + c0 + g * 16 + 8);
      bx1.q[0] = *(const uint4*)(rowB1 + c0 + g * 16);
      bx1.q[1] = *(const uint4*)(rowB1 + c0 + g * 16 + 8);
      bm0.q[0] = *(const uint4*)(rowM0 + c0 + g * 16);
      bm0.q[1] = *(const uint4*)(rowM0 + c0 + g * 16 + 8);
      bm1.q[0] = *(const uint4*)(rowM1 + c0 + g * 16);
      bm1.q[1] = *(const uint4*)(rowM1 + c0 + g * 16 + 8);

      accS[0][0] = wmma_bf16(a0.v, bx0.v, accS[0][0]);
      accS[0][1] = wmma_bf16(a0.v, bx1.v, accS[0][1]);
      accS[1][0] = wmma_bf16(a1.v, bx0.v, accS[1][0]);
      accS[1][1] = wmma_bf16(a1.v, bx1.v, accS[1][1]);
      accM[0][0] = wmma_bf16(a0.v, bm0.v, accM[0][0]);
      accM[0][1] = wmma_bf16(a0.v, bm1.v, accM[0][1]);
      accM[1][0] = wmma_bf16(a1.v, bm0.v, accM[1][0]);
      accM[1][1] = wmma_bf16(a1.v, bm1.v, accM[1][1]);
    }
  }

  // Epilogue: C/D layout -> VGPR r: lanes 0-15 M=r, lanes 16-31 M=r+8; N=lane&15
  #pragma unroll
  for (int f = 0; f < 2; ++f) {
    const int ow   = owBase + f * 16 + fm;
    const int pidx = ((b * H_) + oh) * W_ + ow;
    const float mean = meanArr[pidx];
    const bool  bad  = ncntArr[pidx] >= 576.0f;   // nan_count/1152 >= 0.5
    #pragma unroll
    for (int mi = 0; mi < 2; ++mi) {
      const v8f& S = accS[mi][f];
      const v8f& M = accM[mi][f];
      #pragma unroll
      for (int r = 0; r < 8; ++r) {
        const int oc = ocBase + mi * 16 + g * 8 + r;
        float v = S[r] + mean * M[r] + bias[oc];
        if (bad) v = __builtin_nanf("");
        out[(((size_t)b * OC_ + oc) * H_ + oh) * W_ + ow] = v;
      }
    }
  }
}

// ---------------------------------------------------------------------------
extern "C" void kernel_launch(void* const* d_in, const int* in_sizes, int n_in,
                              void* d_out, int out_size, void* d_ws, size_t ws_size,
                              hipStream_t stream) {
  (void)in_sizes; (void)n_in; (void)out_size; (void)ws_size;
  const float* x    = (const float*)d_in[0];
  const float* kw   = (const float*)d_in[1];
  const float* bias = (const float*)d_in[2];
  float* out = (float*)d_out;

  // Workspace layout (all 256B aligned); total ~143.3 MB.
  char* ws = (char*)d_ws;
  const size_t NPAD = (size_t)B_ * C_ * HP * WP;            // 34,611,200
  __bf16* x0t = (__bf16*)(ws);
  __bf16* mkt = (__bf16*)(ws + NPAD * 2);
  __bf16* Aw  = (__bf16*)(ws + NPAD * 4);
  char*   p   = ws + NPAD * 4 + (size_t)OC_ * KTOT * 2;
  float* csx     = (float*)(p);
  float* csm     = (float*)(p + (size_t)B_ * HP * WP * 4);
  float* meanArr = (float*)(p + (size_t)B_ * HP * WP * 8);
  float* ncntArr = (float*)(p + (size_t)B_ * HP * WP * 8 + (size_t)B_ * H_ * W_ * 4);

  prep_padT  <<<B_ * HP * 4, 256, 0, stream>>>(x, x0t, mkt);
  prep_wt    <<<(OC_ * KTOT + 255) / 256, 256, 0, stream>>>(kw, Aw);
  stats_cs   <<<(B_ * HP * WP + 255) / 256, 256, 0, stream>>>(x0t, mkt, csx, csm);
  stats_patch<<<(B_ * H_ * W_ + 255) / 256, 256, 0, stream>>>(csx, csm, meanArr, ncntArr);

  dim3 grid(2 * H_ * B_, OC_ / 128);   // (4096, 2)
  conv_wmma<<<grid, 256, 0, stream>>>(x0t, mkt, Aw, bias, meanArr, ncntArr, out);
}
